// MultiScaleExtendedMaskedTransformerDecoder_29523605193401
// MI455X (gfx1250) — compile-verified, matrix operand order checked by reading, not run on hardware
//
#include <hip/hip_runtime.h>
#include <hip/hip_bf16.h>

typedef __attribute__((ext_vector_type(16))) __bf16 v16bf;
typedef __attribute__((ext_vector_type(8)))  __bf16 v8bf;
typedef __attribute__((ext_vector_type(2)))  __bf16 v2bf;
typedef __attribute__((ext_vector_type(8)))  float  v8f;
typedef __attribute__((ext_vector_type(2)))  float  v2f;

#define TSEQ 8192
#define DC   768
#define DM   256
#define NB   16
#define NQ   100
#define QPAD 112
#define NH   8
#define HD   32

#if __has_builtin(__builtin_amdgcn_global_load_async_to_lds_b128) && \
    __has_builtin(__builtin_amdgcn_s_wait_asynccnt)
#define HAVE_ASYNC_LDS 1
// exact parameter types per hipcc diagnostic: int4-vector pointers in
// addrspace(1) (global, printed "__device__") and addrspace(3) (LDS).
typedef int v4i_vs __attribute__((vector_size(16)));
typedef __attribute__((address_space(1))) v4i_vs* as1_v4i_p;
typedef __attribute__((address_space(3))) v4i_vs* as3_v4i_p;
#else
#define HAVE_ASYNC_LDS 0
#endif

// Native conversion: ISel picks the packed bf16 cvt.
static __device__ inline v2bf pk2(float x, float y){
  v2f f = {x, y};
  return __builtin_convertvector(f, v2bf);
}

static __device__ inline v16bf zero16(){
  union { v16bf v; unsigned u[8]; } z;
#pragma unroll
  for (int i = 0; i < 8; i++) z.u[i] = 0u;
  return z.v;
}

static __device__ inline v8f wmma_bf16(v16bf a, v16bf b, v8f c){
  return __builtin_amdgcn_wmma_f32_16x16x32_bf16(false, a, false, b, (short)0, c, false, false);
}

// ---------------------------------------------------------------------------
// Conversion kernels (one bandwidth-bound pass)
// ---------------------------------------------------------------------------
__global__ __launch_bounds__(256) void cvt_bf16_kernel(
    const float* __restrict__ src, __bf16* __restrict__ dst, int n8)
{
  int i = blockIdx.x * blockDim.x + threadIdx.x;
  if (i < n8){
    const float4* s = (const float4*)src + 2*(size_t)i;
    float4 a = s[0], b = s[1];
    union { v8bf v; v2bf h[4]; } o;
    o.h[0] = pk2(a.x, a.y); o.h[1] = pk2(a.z, a.w);
    o.h[2] = pk2(b.x, b.y); o.h[3] = pk2(b.z, b.w);
    ((v8bf*)dst)[i] = o.v;
  }
}

__global__ __launch_bounds__(256) void add_cvt_bf16_kernel(
    const float* __restrict__ s0, const float* __restrict__ s1,
    __bf16* __restrict__ dst, int n8)
{
  int i = blockIdx.x * blockDim.x + threadIdx.x;
  if (i < n8){
    const float4* p0 = (const float4*)s0 + 2*(size_t)i;
    const float4* p1 = (const float4*)s1 + 2*(size_t)i;
    float4 a = p0[0], b = p0[1], c = p1[0], d = p1[1];
    union { v8bf v; v2bf h[4]; } o;
    o.h[0] = pk2(a.x + c.x, a.y + c.y); o.h[1] = pk2(a.z + c.z, a.w + c.w);
    o.h[2] = pk2(b.x + d.x, b.y + d.y); o.h[3] = pk2(b.z + d.z, b.w + d.w);
    ((v8bf*)dst)[i] = o.v;
  }
}

// ---------------------------------------------------------------------------
// Kernel 1: K/V projection, M=32 rows/block, fused bias + per-head L2 norm.
//   K -> Kn [B,H,T,hd] bf16 ; V -> Vt [B,H,hd,T] bf16 (transposed over T)
// Wave w owns head w. A-tile (32x32 bf16) staged via async-to-LDS (double
// buffered, ASYNCcnt) and shared by all 8 waves; B tiles software-pipelined
// in registers. 8 WMMAs per k-step.
// ---------------------------------------------------------------------------
#if HAVE_ASYNC_LDS
static __device__ inline void stage_a_async(const __bf16* __restrict__ X,
                                            int b, int t0, int c0,
                                            __bf16* buf, int lane)
{
#pragma unroll
  for (int i = 0; i < 4; i++){
    int row = i*8 + (lane >> 2);
    int col = (lane & 3) * 8;              // 8 bf16 = 16B per lane per op
    const __bf16* g = X + ((size_t)b*TSEQ + t0 + row)*DC + c0 + col;
    __builtin_amdgcn_global_load_async_to_lds_b128(
        (as1_v4i_p)(size_t)(const void*)g,
        (as3_v4i_p)(buf + row*32 + col),
        0, 0);
  }
}
#endif

__global__ __launch_bounds__(256) void kv_proj_kernel(
    const __bf16* __restrict__ X,  const __bf16* __restrict__ Wk,
    const float* __restrict__ bk,  const __bf16* __restrict__ Wv,
    const float* __restrict__ bv,  __bf16* __restrict__ Kn,
    __bf16* __restrict__ Vt)
{
  __shared__ __align__(16) __bf16 abuf[2][32*32];
  __shared__ float smem[32 * DM];
  const int b   = blockIdx.x / (TSEQ/32);
  const int t0  = (blockIdx.x % (TSEQ/32)) * 32;
  const int tid = threadIdx.x;
  const int w   = tid >> 5;
  const int lane= tid & 31;
  const int lo  = lane & 15;
  const int hi  = lane >> 4;

  const __bf16* wk0 = Wk + (size_t)(32*w      + lo)*DC + 16*hi;
  const __bf16* wk1 = Wk + (size_t)(32*w + 16 + lo)*DC + 16*hi;
  const __bf16* wv0 = Wv + (size_t)(32*w      + lo)*DC + 16*hi;
  const __bf16* wv1 = Wv + (size_t)(32*w + 16 + lo)*DC + 16*hi;
#if !HAVE_ASYNC_LDS
  const __bf16* xA = X + ((size_t)b*TSEQ + t0 + lo)*DC + 16*hi;
  const __bf16* xB = xA + (size_t)16*DC;
#endif

  // B pipeline: preload step 0
  v16bf bk0 = *(const v16bf*)wk0;
  v16bf bk1 = *(const v16bf*)wk1;
  v16bf bv0 = *(const v16bf*)wv0;
  v16bf bv1 = *(const v16bf*)wv1;

#if HAVE_ASYNC_LDS
  if (w == 0) stage_a_async(X, b, t0, 0, abuf[0], lane);
  __builtin_amdgcn_s_wait_asynccnt(0);
  __syncthreads();
#endif

  v8f kA0={},kA1={},kB0={},kB1={},vA0={},vA1={},vB0={},vB1={};
#pragma unroll 1
  for (int s = 0; s < DC/32; s++){
    const int c1 = (s+1 < DC/32) ? 32*(s+1) : 32*s;   // clamped next step
#if HAVE_ASYNC_LDS
    if (w == 0 && s+1 < DC/32) stage_a_async(X, b, t0, 32*(s+1), abuf[(s+1)&1], lane);
    const __bf16* ab = abuf[s&1];
    v16bf aA = *(const v16bf*)(ab + lo*32 + 16*hi);
    v16bf aB = *(const v16bf*)(ab + (16 + lo)*32 + 16*hi);
#else
    v16bf aA = *(const v16bf*)(xA + 32*s);
    v16bf aB = *(const v16bf*)(xB + 32*s);
#endif
    // prefetch next B tiles into registers while WMMAs run
    v16bf nk0 = *(const v16bf*)(wk0 + c1);
    v16bf nk1 = *(const v16bf*)(wk1 + c1);
    v16bf nv0 = *(const v16bf*)(wv0 + c1);
    v16bf nv1 = *(const v16bf*)(wv1 + c1);

    kA0 = wmma_bf16(aA, bk0, kA0);  kB0 = wmma_bf16(aB, bk0, kB0);
    kA1 = wmma_bf16(aA, bk1, kA1);  kB1 = wmma_bf16(aB, bk1, kB1);
    vA0 = wmma_bf16(aA, bv0, vA0);  vB0 = wmma_bf16(aB, bv0, vB0);
    vA1 = wmma_bf16(aA, bv1, vA1);  vB1 = wmma_bf16(aB, bv1, vB1);

    bk0 = nk0; bk1 = nk1; bv0 = nv0; bv1 = nv1;
#if HAVE_ASYNC_LDS
    __builtin_amdgcn_s_wait_asynccnt(0);
    __syncthreads();
#endif
  }

  const float bk0s = bk[32*w + lo], bk1s = bk[32*w + 16 + lo];
  const float bv0s = bv[32*w + lo], bv1s = bv[32*w + 16 + lo];
#pragma unroll
  for (int r = 0; r < 8; r++){
    kA0[r]+=bk0s; kA1[r]+=bk1s; kB0[r]+=bk0s; kB1[r]+=bk1s;
    vA0[r]+=bv0s; vA1[r]+=bv1s; vB0[r]+=bv0s; vB1[r]+=bv1s;
  }

  // per-row L2 norm of K over this wave's head (32 cols), both M-tiles
  float ssA[8], ssB[8];
#pragma unroll
  for (int r = 0; r < 8; r++){
    ssA[r] = kA0[r]*kA0[r] + kA1[r]*kA1[r];
    ssB[r] = kB0[r]*kB0[r] + kB1[r]*kB1[r];
  }
#pragma unroll
  for (int r = 0; r < 8; r++){
    for (int off = 1; off < 16; off <<= 1){
      ssA[r] += __shfl_xor(ssA[r], off, 32);
      ssB[r] += __shfl_xor(ssB[r], off, 32);
    }
    float iA = 1.0f / fmaxf(sqrtf(ssA[r]), 1e-6f);
    float iB = 1.0f / fmaxf(sqrtf(ssB[r]), 1e-6f);
    kA0[r]*=iA; kA1[r]*=iA; kB0[r]*=iB; kB1[r]*=iB;
  }

  // stage K (32 rows x 256 cols), emit packed 64B rows of Kn[b,h,t,:]
#pragma unroll
  for (int r = 0; r < 8; r++){
    smem[(r + 8*hi)*DM + 32*w + lo]           = kA0[r];
    smem[(r + 8*hi)*DM + 32*w + 16 + lo]      = kA1[r];
    smem[(16 + r + 8*hi)*DM + 32*w + lo]      = kB0[r];
    smem[(16 + r + 8*hi)*DM + 32*w + 16 + lo] = kB1[r];
  }
  __syncthreads();
  {
    const float* src = smem + lane*DM + 32*w;
    union { v16bf v; v2bf h[8]; } o1, o2;
#pragma unroll
    for (int i = 0; i < 8; i++){
      o1.h[i] = pk2(src[2*i],      src[2*i+1]);
      o2.h[i] = pk2(src[16 + 2*i], src[16 + 2*i+1]);
    }
    __bf16* dst = Kn + (((size_t)b*NH + w)*TSEQ + t0 + lane)*HD;
    *(v16bf*)dst        = o1.v;
    *(v16bf*)(dst + 16) = o2.v;
  }
  __syncthreads();
  // stage V, emit transposed Vt[b,h,d,t0..t0+32)
#pragma unroll
  for (int r = 0; r < 8; r++){
    smem[(r + 8*hi)*DM + 32*w + lo]           = vA0[r];
    smem[(r + 8*hi)*DM + 32*w + 16 + lo]      = vA1[r];
    smem[(16 + r + 8*hi)*DM + 32*w + lo]      = vB0[r];
    smem[(16 + r + 8*hi)*DM + 32*w + 16 + lo] = vB1[r];
  }
  __syncthreads();
  {
    const int d = lane;
    union { v16bf v; v2bf h[8]; } p1, p2;
#pragma unroll
    for (int i = 0; i < 8; i++){
      p1.h[i] = pk2(smem[(2*i)*DM + 32*w + d],      smem[(2*i+1)*DM + 32*w + d]);
      p2.h[i] = pk2(smem[(16 + 2*i)*DM + 32*w + d], smem[(16 + 2*i+1)*DM + 32*w + d]);
    }
    __bf16* dst = Vt + (((size_t)b*NH + w)*HD + d)*TSEQ + t0;
    *(v16bf*)dst        = p1.v;
    *(v16bf*)(dst + 16) = p2.v;
  }
}

// ---------------------------------------------------------------------------
// Kernel 2: q projection from pre-added (tgt+query_pos) bf16, fused bias +
// per-head L2-norm + fold scale*(1-alpha).  -> Qn [B,H,QPAD,hd] bf16
// ---------------------------------------------------------------------------
__global__ __launch_bounds__(256) void q_proj_kernel(
    const __bf16* __restrict__ Qin, const __bf16* __restrict__ Wq,
    const float* __restrict__ bq,   const float* __restrict__ log_scale,
    __bf16* __restrict__ Qn)
{
  __shared__ float smem[16 * DM];
  const int b   = blockIdx.x / (QPAD/16);
  const int q0  = (blockIdx.x % (QPAD/16)) * 16;
  const int tid = threadIdx.x;
  const int w   = tid >> 5;
  const int lane= tid & 31;
  const int lo  = lane & 15;
  const int hi  = lane >> 4;
  const int q   = q0 + lo;
  const bool valid = (q < NQ);

  const __bf16* qr = Qin + ((size_t)q*NB + b)*DM + 16*hi;
  const __bf16* w0 = Wq + (size_t)(32*w      + lo)*DM + 16*hi;
  const __bf16* w1 = Wq + (size_t)(32*w + 16 + lo)*DM + 16*hi;

  v8f c0a = {}, c1a = {};
#pragma unroll 1
  for (int c0 = 0; c0 < DM; c0 += 32){
    v16bf a  = valid ? *(const v16bf*)(qr + c0) : zero16();
    v16bf b0 = *(const v16bf*)(w0 + c0);
    v16bf b1 = *(const v16bf*)(w1 + c0);
    c0a = wmma_bf16(a, b0, c0a);
    c1a = wmma_bf16(a, b1, c1a);
  }
  const float b0s = bq[32*w + lo], b1s = bq[32*w + 16 + lo];
#pragma unroll
  for (int r = 0; r < 8; r++){ c0a[r]+=b0s; c1a[r]+=b1s; }

  const float sc = fminf(__expf(log_scale[0]), 100.0f) * (1.0f - (float)w / 7.0f);
  float ss[8];
#pragma unroll
  for (int r = 0; r < 8; r++) ss[r] = c0a[r]*c0a[r] + c1a[r]*c1a[r];
#pragma unroll
  for (int r = 0; r < 8; r++){
    for (int off = 1; off < 16; off <<= 1) ss[r] += __shfl_xor(ss[r], off, 32);
    float inv = sc / fmaxf(sqrtf(ss[r]), 1e-6f);
    c0a[r] *= inv; c1a[r] *= inv;
  }
#pragma unroll
  for (int r = 0; r < 8; r++){
    smem[(r + 8*hi)*DM + 32*w + lo]      = c0a[r];
    smem[(r + 8*hi)*DM + 32*w + 16 + lo] = c1a[r];
  }
  __syncthreads();
  {
    int row = lane >> 1, dlo = (lane & 1) * 16;
    const float* src = smem + row*DM + 32*w + dlo;
    union { v16bf v; v2bf h[8]; } o;
#pragma unroll
    for (int i = 0; i < 8; i++) o.h[i] = pk2(src[2*i], src[2*i+1]);
    __bf16* dst = Qn + (((size_t)b*NH + w)*QPAD + q0 + row)*HD + dlo;
    *(v16bf*)dst = o.v;
  }
}

// ---------------------------------------------------------------------------
// Kernel 3: fused flash attention with per-head oov blending.
// Block = (b, q-tile of 16); wave w == head w -> oov cachelines shared in-WGP.
// ---------------------------------------------------------------------------
__global__ __launch_bounds__(256) void attn_kernel(
    const __bf16* __restrict__ Qn, const __bf16* __restrict__ Kn,
    const __bf16* __restrict__ Vt, const float* __restrict__ oov,
    __bf16* __restrict__ AO)
{
  const int b   = blockIdx.x / (QPAD/16);
  const int q0  = (blockIdx.x % (QPAD/16)) * 16;
  const int tid = threadIdx.x;
  const int h   = tid >> 5;
  const int lane= tid & 31;
  const int lo  = lane & 15;
  const int hi  = lane >> 4;
  const int qrow= q0 + lo;
  const int qc  = qrow < NQ ? qrow : NQ - 1;
  const float alpha = (float)h / 7.0f;

  const v16bf qB = *(const v16bf*)(Qn + (((size_t)b*NH + h)*QPAD + qrow)*HD + 16*hi);
  const __bf16* Kb = Kn + ((size_t)b*NH + h) * (size_t)TSEQ * HD;
  const __bf16* Vb = Vt + ((size_t)b*NH + h) * (size_t)HD * TSEQ;
  const float*  op = oov + ((size_t)b*NQ + qc) * TSEQ;

  float mrun = -3.0e38f, lrun = 0.0f;
  v8f acc0 = {}, acc1 = {};

#pragma unroll 1
  for (int t0 = 0; t0 < TSEQ; t0 += 32){
    // stream-ahead prefetch (global_prefetch_b8), 128 keys ahead
    {
      int tp = t0 + 128;
      if (tp < TSEQ){
        __builtin_prefetch(Kb + (size_t)(tp + lo)*HD, 0, 1);
        __builtin_prefetch(Vb + (size_t)lo*TSEQ + tp, 0, 1);
        __builtin_prefetch(op + tp, 0, 1);
      }
    }
    v16bf a0 = *(const v16bf*)(Kb + (size_t)(t0      + lo)*HD + 16*hi);
    v16bf a1 = *(const v16bf*)(Kb + (size_t)(t0 + 16 + lo)*HD + 16*hi);
    v8f s0 = {}, s1 = {};
    s0 = wmma_bf16(a0, qB, s0);          // S^T[key 0..15][query]
    s1 = wmma_bf16(a1, qB, s1);          // S^T[key16..31][query]

    union { float4 q4[2]; float f[8]; } ov0, ov1;
    ov0.q4[0] = ((const float4*)(op + t0      + 8*hi))[0];
    ov0.q4[1] = ((const float4*)(op + t0      + 8*hi))[1];
    ov1.q4[0] = ((const float4*)(op + t0 + 16 + 8*hi))[0];
    ov1.q4[1] = ((const float4*)(op + t0 + 16 + 8*hi))[1];

    float z0[8], z1[8], mc = -3.0e38f;
#pragma unroll
    for (int r = 0; r < 8; r++){
      z0[r] = s0[r] + alpha * ov0.f[r];
      z1[r] = s1[r] + alpha * ov1.f[r];
      mc = fmaxf(mc, fmaxf(z0[r], z1[r]));
    }
    mc = fmaxf(mc, __shfl_xor(mc, 16, 32));
    const float mn   = fmaxf(mrun, mc);
    const float corr = __expf(mrun - mn);
    mrun = mn;

    float p0[8], p1[8], s = 0.0f;
#pragma unroll
    for (int r = 0; r < 8; r++){
      p0[r] = __expf(z0[r] - mn);
      p1[r] = __expf(z1[r] - mn);
      s += p0[r] + p1[r];
    }
    lrun = lrun * corr + s;
#pragma unroll
    for (int r = 0; r < 8; r++){ acc0[r] *= corr; acc1[r] *= corr; }

    // half-wave swap so each lane holds its B-layout key range (P^T as B)
    float rcv[8];
#pragma unroll
    for (int r = 0; r < 8; r++){
      float snd = hi ? p0[r] : p1[r];
      rcv[r] = __shfl_xor(snd, 16, 32);
    }
    union { v16bf v; v2bf h2[8]; } pb;
#pragma unroll
    for (int j = 0; j < 4; j++){
      float a0_ = hi ? rcv[2*j]   : p0[2*j];
      float a1_ = hi ? rcv[2*j+1] : p0[2*j+1];
      float b0_ = hi ? p1[2*j]    : rcv[2*j];
      float b1_ = hi ? p1[2*j+1]  : rcv[2*j+1];
      pb.h2[j]     = pk2(a0_, a1_);
      pb.h2[4 + j] = pk2(b0_, b1_);
    }

    v16bf va0 = *(const v16bf*)(Vb + (size_t)(lo)     *TSEQ + t0 + 16*hi);
    v16bf va1 = *(const v16bf*)(Vb + (size_t)(16 + lo)*TSEQ + t0 + 16*hi);
    acc0 = wmma_bf16(va0, pb.v, acc0);   // O^T[d 0..15][query]
    acc1 = wmma_bf16(va1, pb.v, acc1);   // O^T[d16..31][query]
  }

  const float ltot = lrun + __shfl_xor(lrun, 16, 32);
  const float inv  = 1.0f / ltot;
  if (qrow < NQ){
    union { v8bf v; v2bf h2[4]; } o0, o1;
#pragma unroll
    for (int j = 0; j < 4; j++){
      o0.h2[j] = pk2(acc0[2*j]*inv, acc0[2*j+1]*inv);
      o1.h2[j] = pk2(acc1[2*j]*inv, acc1[2*j+1]*inv);
    }
    __bf16* base = AO + ((size_t)qrow*NB + b)*DM + h*HD;
    *(v8bf*)(base + 8*hi)      = o0.v;
    *(v8bf*)(base + 16 + 8*hi) = o1.v;
  }
}

// ---------------------------------------------------------------------------
// Kernel 4: output projection + residual + LayerNorm -> d_out [Q,B,D] f32
// ---------------------------------------------------------------------------
__global__ __launch_bounds__(256) void out_ln_kernel(
    const __bf16* __restrict__ AO, const __bf16* __restrict__ Wo,
    const float* __restrict__ bo,  const float* __restrict__ tgt,
    const float* __restrict__ g,   const float* __restrict__ beta,
    float* __restrict__ out)
{
  __shared__ float xs[16 * DM];
  __shared__ float red1[256], red2[256];
  __shared__ float mu_s[16], rs_s[16];
  const int row0 = blockIdx.x * 16;
  const int tid  = threadIdx.x;
  const int w    = tid >> 5;
  const int lane = tid & 31;
  const int lo   = lane & 15;
  const int hi   = lane >> 4;

  const __bf16* ar = AO + (size_t)(row0 + lo)*DM + 16*hi;
  const __bf16* w0 = Wo + (size_t)(32*w      + lo)*DM + 16*hi;
  const __bf16* w1 = Wo + (size_t)(32*w + 16 + lo)*DM + 16*hi;

  v8f c0a = {}, c1a = {};
#pragma unroll 1
  for (int c0 = 0; c0 < DM; c0 += 32){
    v16bf a  = *(const v16bf*)(ar + c0);
    v16bf b0 = *(const v16bf*)(w0 + c0);
    v16bf b1 = *(const v16bf*)(w1 + c0);
    c0a = wmma_bf16(a, b0, c0a);
    c1a = wmma_bf16(a, b1, c1a);
  }
  const float b0s = bo[32*w + lo], b1s = bo[32*w + 16 + lo];
#pragma unroll
  for (int r = 0; r < 8; r++){
    int row = row0 + r + 8*hi;
    float x0 = c0a[r] + b0s + tgt[(size_t)row*DM + 32*w + lo];
    float x1 = c1a[r] + b1s + tgt[(size_t)row*DM + 32*w + 16 + lo];
    xs[(r + 8*hi)*DM + 32*w + lo]      = x0;
    xs[(r + 8*hi)*DM + 32*w + 16 + lo] = x1;
  }
  __syncthreads();
  {
    int row = tid >> 4, ch = tid & 15;
    float s1 = 0.0f, s2 = 0.0f;
    const float* p = xs + row*DM + ch*16;
#pragma unroll
    for (int i = 0; i < 16; i++){ float v = p[i]; s1 += v; s2 += v*v; }
    red1[tid] = s1; red2[tid] = s2;
  }
  __syncthreads();
  if (tid < 16){
    float s1 = 0.0f, s2 = 0.0f;
#pragma unroll
    for (int i = 0; i < 16; i++){ s1 += red1[tid*16 + i]; s2 += red2[tid*16 + i]; }
    float mu  = s1 / (float)DM;
    float var = s2 / (float)DM - mu*mu;
    mu_s[tid] = mu;
    rs_s[tid] = rsqrtf(var + 1e-5f);
  }
  __syncthreads();
  {
    int row = tid >> 4, ch = tid & 15;
    float mu = mu_s[row], rs = rs_s[row];
    float*       po = out + (size_t)(row0 + row)*DM + ch*16;
    const float* px = xs + row*DM + ch*16;
#pragma unroll
    for (int i = 0; i < 16; i++){
      int e = ch*16 + i;
      po[i] = (px[i] - mu) * rs * g[e] + beta[e];
    }
  }
}

extern "C" void kernel_launch(void* const* d_in, const int* in_sizes, int n_in,
                              void* d_out, int out_size, void* d_ws, size_t ws_size,
                              hipStream_t stream)
{
  const float* tgt = (const float*)d_in[0];
  const float* txt = (const float*)d_in[1];
  const float* qps = (const float*)d_in[2];
  const float* oov = (const float*)d_in[3];
  const float* Wq  = (const float*)d_in[4];
  const float* bq  = (const float*)d_in[5];
  const float* Wk  = (const float*)d_in[6];
  const float* bk  = (const float*)d_in[7];
  const float* Wv  = (const float*)d_in[8];
  const float* bv  = (const float*)d_in[9];
  const float* Wo  = (const float*)d_in[10];
  const float* bo  = (const float*)d_in[11];
  const float* lng = (const float*)d_in[12];
  const float* lnb = (const float*)d_in[13];
  const float* lsc = (const float*)d_in[14];

  // workspace layout (bytes)
  char* ws = (char*)d_ws;
  const size_t SZ_XBF = (size_t)NB*TSEQ*DC*2;            // 201,326,592
  const size_t SZ_KN  = (size_t)NB*NH*TSEQ*HD*2;         //  67,108,864
  const size_t SZ_VT  = SZ_KN;                           //  67,108,864
  const size_t SZ_QN  = (size_t)NB*NH*QPAD*HD*2;         //     917,504
  const size_t SZ_QIN = (size_t)NQ*NB*DM*2;              //     819,200
  const size_t SZ_WKV = (size_t)DM*DC*2;                 //     393,216
  const size_t SZ_WSM = (size_t)DM*DM*2;                 //     131,072

  __bf16* Xbf = (__bf16*)(ws);
  __bf16* Kn  = (__bf16*)(ws + SZ_XBF);
  __bf16* Vt  = (__bf16*)(ws + SZ_XBF + SZ_KN);
  __bf16* Qn  = (__bf16*)(ws + SZ_XBF + SZ_KN + SZ_VT);
  __bf16* Qin = (__bf16*)(ws + SZ_XBF + SZ_KN + SZ_VT + SZ_QN);
  __bf16* Wkb = (__bf16*)(ws + SZ_XBF + SZ_KN + SZ_VT + SZ_QN + SZ_QIN);
  __bf16* Wvb = (__bf16*)(ws + SZ_XBF + SZ_KN + SZ_VT + SZ_QN + SZ_QIN + SZ_WKV);
  __bf16* Wqb = (__bf16*)(ws + SZ_XBF + SZ_KN + SZ_VT + SZ_QN + SZ_QIN + 2*SZ_WKV);
  __bf16* Wob = (__bf16*)(ws + SZ_XBF + SZ_KN + SZ_VT + SZ_QN + SZ_QIN + 2*SZ_WKV + SZ_WSM);
  __bf16* AO  = (__bf16*)(ws + SZ_XBF + SZ_KN + SZ_VT + SZ_QN + SZ_QIN + 2*SZ_WKV + 2*SZ_WSM);

  // pre-convert operands to bf16
  {
    int n8 = (NB*TSEQ*DC)/8;
    cvt_bf16_kernel<<<(n8 + 255)/256, 256, 0, stream>>>(txt, Xbf, n8);
  }
  {
    int n8 = (DM*DC)/8;
    cvt_bf16_kernel<<<(n8 + 255)/256, 256, 0, stream>>>(Wk, Wkb, n8);
    cvt_bf16_kernel<<<(n8 + 255)/256, 256, 0, stream>>>(Wv, Wvb, n8);
  }
  {
    int n8 = (DM*DM)/8;
    cvt_bf16_kernel<<<(n8 + 255)/256, 256, 0, stream>>>(Wq, Wqb, n8);
    cvt_bf16_kernel<<<(n8 + 255)/256, 256, 0, stream>>>(Wo, Wob, n8);
  }
  {
    int n8 = (NQ*NB*DM)/8;
    add_cvt_bf16_kernel<<<(n8 + 255)/256, 256, 0, stream>>>(tgt, qps, Qin, n8);
  }

  q_proj_kernel<<<NB * (QPAD/16), 256, 0, stream>>>(Qin, Wqb, bq, lsc, Qn);
  kv_proj_kernel<<<NB * (TSEQ/32), 256, 0, stream>>>(Xbf, Wkb, bk, Wvb, bv, Kn, Vt);
  attn_kernel<<<NB * (QPAD/16), 256, 0, stream>>>(Qn, Kn, Vt, oov, AO);
  out_ln_kernel<<<(NQ * NB) / 16, 256, 0, stream>>>(AO, Wob, bo, tgt, lng, lnb,
                                                    (float*)d_out);
}
